// QuanvolutionGen216_65481071406272
// MI455X (gfx1250) — matrix operand order repaired and checked by reading, not compiled, not run on hardware
//
#include <hip/hip_runtime.h>

typedef _Float16 v8h  __attribute__((ext_vector_type(8)));
typedef _Float16 v16h __attribute__((ext_vector_type(16)));
typedef float    v8f  __attribute__((ext_vector_type(8)));

#define B_N     1024
#define P_N     196
#define KDIM    784
#define KPAD    800   // 25 * 32
#define NCLS    10

// ---------------------------------------------------------------------------
// Kernel 1: quantum circuit simulation, one thread per (batch, patch).
// State: 16 complex amplitudes in registers, index bits [w0 w1 w2 w3], w0=bit3.
// Writes feats as f16 into [B_N, KPAD]; p==0 threads zero the K tail.
// ---------------------------------------------------------------------------
__global__ void __launch_bounds__(256)
quanv_kernel(const float* __restrict__ x,
             const float* __restrict__ qp,
             _Float16* __restrict__ feats)
{
    int idx = blockIdx.x * blockDim.x + threadIdx.x;
    if (idx >= B_N * P_N) return;
    int b = idx / P_N, p = idx - b * P_N;
    int pr = p / 14, pc = p - pr * 14;

    const float* xb = x + b * 784 + (pr * 2) * 28 + pc * 2;
    float ang[4] = { xb[0], xb[1], xb[28], xb[29] };   // [tl,tr,bl,br] = wires 0..3

    float cg[4], sg[4];
#pragma unroll
    for (int w = 0; w < 4; ++w) __sincosf(ang[w] * 0.5f, &sg[w], &cg[w]);

    float re[16], im[16];
#pragma unroll
    for (int i = 0; i < 16; ++i) { re[i] = 0.f; im[i] = 0.f; }
    re[0] = 1.0f;

#pragma unroll
    for (int layer = 0; layer < 3; ++layer) {
        // RX(x[w]) on each wire: [[c, -i s],[-i s, c]]
#pragma unroll
        for (int w = 0; w < 4; ++w) {
            const int bw = 8 >> w;
            float c = cg[w], s = sg[w];
#pragma unroll
            for (int i = 0; i < 16; ++i) {
                if (i & bw) continue;
                int j = i | bw;
                float r0 = re[i], i0 = im[i], r1 = re[j], i1 = im[j];
                re[i] = c * r0 + s * i1;  im[i] = c * i0 - s * r1;
                re[j] = c * r1 + s * i0;  im[j] = c * i1 - s * r0;
            }
        }
        // CNOT(0,1): ctrl bit 8, tgt bit 4
#pragma unroll
        for (int i = 0; i < 16; ++i) {
            if ((i & 8) && !(i & 4)) {
                int j = i | 4;
                float tr = re[i], ti = im[i];
                re[i] = re[j]; im[i] = im[j]; re[j] = tr; im[j] = ti;
            }
        }
        // CNOT(2,3): ctrl bit 2, tgt bit 1
#pragma unroll
        for (int i = 0; i < 16; ++i) {
            if ((i & 2) && !(i & 1)) {
                int j = i | 1;
                float tr = re[i], ti = im[i];
                re[i] = re[j]; im[i] = im[j]; re[j] = tr; im[j] = ti;
            }
        }
        // RZ(params[layer][w]): diag(e^{-i phi/2}, e^{+i phi/2})
#pragma unroll
        for (int w = 0; w < 4; ++w) {
            const int bw = 8 >> w;
            float sp, cp;
            __sincosf(qp[layer * 4 + w] * 0.5f, &sp, &cp);
#pragma unroll
            for (int i = 0; i < 16; ++i) {
                float r = re[i], q = im[i];
                if (i & bw) { re[i] = cp * r - sp * q; im[i] = cp * q + sp * r; }
                else        { re[i] = cp * r + sp * q; im[i] = cp * q - sp * r; }
            }
        }
    }

    float prob[16];
#pragma unroll
    for (int i = 0; i < 16; ++i) prob[i] = re[i] * re[i] + im[i] * im[i];

    _Float16* out = feats + (size_t)b * KPAD + p * 4;
#pragma unroll
    for (int w = 0; w < 4; ++w) {
        const int bw = 8 >> w;
        float z = 0.f;
#pragma unroll
        for (int i = 0; i < 16; ++i) z += (i & bw) ? -prob[i] : prob[i];
        out[w] = (_Float16)z;
    }

    // zero-pad K tail (784..799) once per batch row
    if (p == 0) {
        _Float16* tail = feats + (size_t)b * KPAD + KDIM;
#pragma unroll
        for (int t = 0; t < KPAD - KDIM; ++t) tail[t] = (_Float16)0.f;
    }
}

// ---------------------------------------------------------------------------
// Kernel 2: repack W [10,784] f32 -> Wh [16,800] f16 (zero padded rows/cols)
// ---------------------------------------------------------------------------
__global__ void __launch_bounds__(256)
pack_w_kernel(const float* __restrict__ W, _Float16* __restrict__ Wh)
{
    int i = blockIdx.x * blockDim.x + threadIdx.x;
    if (i >= 16 * KPAD) return;
    int n = i / KPAD, k = i - n * KPAD;
    float v = (n < NCLS && k < KDIM) ? W[n * KDIM + k] : 0.f;
    Wh[i] = (_Float16)v;
}

// ---------------------------------------------------------------------------
// Kernel 3: WMMA head. One wave per 16-row tile of the output.
// C[1024,16] = feats[1024,800] @ Wh^T, then bias + log_softmax over cols 0..9.
// A frag (16x32 f16): lane L: M=L&15; halves 0-15 -> K = k0 + (L<16?0:16) + h
// B frag (32x16 f16): lane L: N=L&15; halves 0-15 -> K = k0 + (L<16?0:16) + h
// C/D (v8f): elem e -> row (e + (L<16?0:8)), col L&15
// ---------------------------------------------------------------------------
__global__ void __launch_bounds__(128)
head_kernel(const _Float16* __restrict__ feats,
            const _Float16* __restrict__ Wh,
            const float* __restrict__ bias,
            float* __restrict__ out)
{
    __shared__ float lds[4][16 * 16];

    int lane = threadIdx.x & 31;
    int wave = threadIdx.x >> 5;
    int tile = blockIdx.x * 4 + wave;          // 0..63

    int mcol  = lane & 15;
    int koffA = (lane < 16) ? 0 : 8;
    int koffB = (lane < 16) ? 0 : 16;

    const _Float16* arow = feats + (size_t)(tile * 16 + mcol) * KPAD;
    const _Float16* brow = Wh + (size_t)mcol * KPAD + koffB;

    v8f acc = {};
#pragma unroll 5
    for (int kt = 0; kt < KPAD / 32; ++kt) {
        int k0 = kt * 32;
        v16h a;
        a = *(const v16h*)(arow + k0 + koffB);
        v16h bm = *(const v16h*)(brow + k0);
        acc = __builtin_amdgcn_wmma_f32_16x16x32_f16(
                  false, a, false, bm, (short)0, acc, false, false);
    }

    int rbase = (lane < 16) ? 0 : 8;
#pragma unroll
    for (int e = 0; e < 8; ++e)
        lds[wave][(rbase + e) * 16 + mcol] = acc[e];
    __syncthreads();

    if (lane < 16) {
        float l[NCLS];
        float mx = -1e30f;
#pragma unroll
        for (int n = 0; n < NCLS; ++n) {
            l[n] = lds[wave][lane * 16 + n] + bias[n];
            mx = fmaxf(mx, l[n]);
        }
        float ssum = 0.f;
#pragma unroll
        for (int n = 0; n < NCLS; ++n) ssum += __expf(l[n] - mx);
        float lse = mx + __logf(ssum);
        float* orow = out + (size_t)(tile * 16 + lane) * NCLS;
#pragma unroll
        for (int n = 0; n < NCLS; ++n) orow[n] = l[n] - lse;
    }
}

// ---------------------------------------------------------------------------
extern "C" void kernel_launch(void* const* d_in, const int* in_sizes, int n_in,
                              void* d_out, int out_size, void* d_ws, size_t ws_size,
                              hipStream_t stream)
{
    const float* x  = (const float*)d_in[0];   // [1024,1,28,28]
    const float* qp = (const float*)d_in[1];   // [3,4]
    const float* W  = (const float*)d_in[2];   // [10,784]
    const float* bb = (const float*)d_in[3];   // [10]
    float* out = (float*)d_out;                // [1024,10]

    _Float16* feats = (_Float16*)d_ws;                              // 1024*800*2 B
    _Float16* Wh    = (_Float16*)((char*)d_ws + (size_t)B_N * KPAD * 2);  // 16*800*2 B

    quanv_kernel<<<(B_N * P_N + 255) / 256, 256, 0, stream>>>(x, qp, feats);
    pack_w_kernel<<<(16 * KPAD + 255) / 256, 256, 0, stream>>>(W, Wh);
    head_kernel<<<16, 128, 0, stream>>>(feats, Wh, bb, out);
}
